// LocationSensitiveAttention_23708219474640
// MI455X (gfx1250) — matrix operand order repaired and checked by reading, not compiled, not run on hardware
//
#include <hip/hip_runtime.h>

typedef __attribute__((ext_vector_type(16))) _Float16 v16h;
typedef __attribute__((ext_vector_type(8)))  float    v8f;

#define B_ 128
#define T_ 4096
#define U_ 128
#define F_ 32
#define KW 31
#define TBLOCK 512
#define NTHREADS 256
#define NWAVES 8
#define NEGV (-4294967295.0f)

#if __has_builtin(__builtin_amdgcn_tanhf)
#  define TANHF(x) __builtin_amdgcn_tanhf(x)
#elif __has_builtin(__builtin_amdgcn_tanh_f32)
#  define TANHF(x) __builtin_amdgcn_tanh_f32(x)
#else
#  define TANHF(x) tanhf(x)
#endif

// ---------------------------------------------------------------------------
// Kernel 1: fused conv(31) -> GEMM(32x128) -> tanh-reduce energy, one pass
// over the 256MB `memory` tensor. Both matmuls use v_wmma_f32_16x16x32_f16.
// ---------------------------------------------------------------------------
__global__ __launch_bounds__(NTHREADS)
void lsa_energy_kernel(const float* __restrict__ query,
                       const float* __restrict__ state,
                       const float* __restrict__ memory,
                       const float* __restrict__ conv_w,
                       const float* __restrict__ conv_b,
                       const float* __restrict__ loc_w,
                       const float* __restrict__ v_a,
                       const float* __restrict__ b_a,
                       float* __restrict__ energy)
{
    __shared__ float    s_state[TBLOCK + 48];       // state window (zero padded)
    __shared__ _Float16 s_wconv[32][33];            // conv weights [k][f], k=31 zero
    __shared__ _Float16 s_locw[32][129];            // loc_w [f][u]
    __shared__ float    s_qb[U_];                   // query[b,:] + b_a
    __shared__ float    s_va[U_];
    __shared__ _Float16 s_stage[NWAVES][16][33];    // wave-private C->A relayout

    const int b    = blockIdx.x;
    const int tblk = blockIdx.y * TBLOCK;
    const int tid  = threadIdx.x;
    const int lane = tid & 31;
    const int wave = tid >> 5;
    const int hlf  = lane >> 4;    // 0: lanes 0-15, 1: lanes 16-31
    const int nl   = lane & 15;

    // ---- cooperative LDS fills ------------------------------------------
    for (int i = tid; i < TBLOCK + 31; i += NTHREADS) {
        int t = tblk - 15 + i;
        s_state[i] = (t >= 0 && t < T_) ? state[(size_t)b * T_ + t] : 0.0f;
    }
    for (int i = tid; i < 32 * 32; i += NTHREADS) {
        int k = i >> 5, f = i & 31;
        s_wconv[k][f] = (k < KW) ? (_Float16)conv_w[k * F_ + f] : (_Float16)0.0f;
    }
    for (int i = tid; i < F_ * U_; i += NTHREADS) {
        int f = i >> 7, u = i & 127;
        s_locw[f][u] = (_Float16)loc_w[i];
    }
    if (tid < U_) {
        s_qb[tid] = query[(size_t)b * U_ + tid] + b_a[tid];
        s_va[tid] = v_a[tid];
    }
    __syncthreads();

    // ---- register-resident B tiles (ISA dense 16-bit B layout:
    //      lane = column N (l&15); K = 8*(l>>4) + i + 8*(i>>3)) -------------
    v16h Bconv[2];
    #pragma unroll
    for (int tf = 0; tf < 2; ++tf)
        #pragma unroll
        for (int i = 0; i < 16; ++i)
            Bconv[tf][i] = s_wconv[hlf * 8 + i + ((i >> 3) << 3)][tf * 16 + nl];

    v16h Blw[8];
    #pragma unroll
    for (int tu = 0; tu < 8; ++tu)
        #pragma unroll
        for (int i = 0; i < 16; ++i)
            Blw[tu][i] = s_locw[hlf * 8 + i + ((i >> 3) << 3)][tu * 16 + nl];

    float qb[8], va[8];
    #pragma unroll
    for (int tu = 0; tu < 8; ++tu) {
        qb[tu] = s_qb[tu * 16 + nl];
        va[tu] = s_va[tu * 16 + nl];
    }
    const float cb0 = conv_b[nl];
    const float cb1 = conv_b[16 + nl];

    // ---- per-wave 16-row t tiles ----------------------------------------
    for (int tt = wave; tt < TBLOCK / 16; tt += NWAVES) {
        const int t0 = tblk + tt * 16;

        // conv as WMMA: A[t][k] = state[t0 + t + k - 15]  (band matrix)
        // dense 16-bit A swizzle: VGPR v holds K pair (Kv, Kv+1),
        // Kv = 2*(v&3) + 16*(v>>2) + 8*(lane>>4)
        v16h Ac;
        const int base = (t0 - tblk) + nl;   // s_state index of k=0, row nl
        #pragma unroll
        for (int v = 0; v < 8; ++v) {
            const int K = ((v & 3) << 1) + ((v >> 2) << 4) + (hlf << 3);
            Ac[2 * v]     = (_Float16)s_state[base + K];
            Ac[2 * v + 1] = (_Float16)s_state[base + K + 1];
        }
        v8f Cf0 = {}, Cf1 = {};
        Cf0 = __builtin_amdgcn_wmma_f32_16x16x32_f16(false, Ac, false, Bconv[0],
                                                     (short)0, Cf0, false, false);
        Cf1 = __builtin_amdgcn_wmma_f32_16x16x32_f16(false, Ac, false, Bconv[1],
                                                     (short)0, Cf1, false, false);

        // stage loc_feat: f32 C layout -> f16 A layout (wave-private LDS,
        // per-wave DS ops are in-order so no barrier is required)
        #pragma unroll
        for (int r = 0; r < 8; ++r) {
            s_stage[wave][r + 8 * hlf][nl]      = (_Float16)(Cf0[r] + cb0);
            s_stage[wave][r + 8 * hlf][16 + nl] = (_Float16)(Cf1[r] + cb1);
        }
        v16h Al;
        #pragma unroll
        for (int v = 0; v < 8; ++v) {
            const int K = ((v & 3) << 1) + ((v >> 2) << 4) + (hlf << 3);
            Al[2 * v]     = s_stage[wave][nl][K];
            Al[2 * v + 1] = s_stage[wave][nl][K + 1];
        }

        // proc_loc tile (16t x 128u) fully in accumulators: 8 WMMAs
        v8f C[8];
        #pragma unroll
        for (int tu = 0; tu < 8; ++tu) {
            v8f cc = {};
            cc = __builtin_amdgcn_wmma_f32_16x16x32_f16(false, Al, false, Blw[tu],
                                                        (short)0, cc, false, false);
            C[tu] = cc;
        }

        // energy: e[t] = sum_u va[u] * tanh(mem + qb + proc_loc)
        // C layout: (vgpr r, lane) -> (t_local = r + 8*hlf, u = 16*tu + nl)
        const float* mrow = memory + ((size_t)b * T_ + t0) * U_;
        float acc[8];
        #pragma unroll
        for (int r = 0; r < 8; ++r) acc[r] = 0.0f;

        #pragma unroll
        for (int tu = 0; tu < 8; ++tu) {
            const int u = tu * 16 + nl;
            #pragma unroll
            for (int r = 0; r < 8; ++r) {
                const int tl = r + 8 * hlf;
                float x = mrow[tl * U_ + u] + qb[tu] + C[tu][r];
                acc[r] += va[tu] * TANHF(x);
            }
        }
        // reduce over the 16 u-lanes of each half-wave
        #pragma unroll
        for (int r = 0; r < 8; ++r) {
            float vsum = acc[r];
            vsum += __shfl_xor(vsum, 1);
            vsum += __shfl_xor(vsum, 2);
            vsum += __shfl_xor(vsum, 4);
            vsum += __shfl_xor(vsum, 8);
            if (nl == 0)
                energy[(size_t)b * T_ + t0 + r + 8 * hlf] = vsum;
        }
    }
}

// ---------------------------------------------------------------------------
// Kernel 2: window mask + softmax + argmax + next_state (per row b).
// Energy lives in the alignments region of d_out and is overwritten in place.
// Keep window: t in [pma-4, pma+2]  (WIN=7).
// ---------------------------------------------------------------------------
__global__ __launch_bounds__(NTHREADS)
void lsa_softmax_kernel(const float* __restrict__ state,
                        const int* __restrict__ pma,
                        float* __restrict__ alignments,   // in: energy, out: alignments
                        float* __restrict__ next_state,
                        int* __restrict__ max_att)
{
    __shared__ float s_red[NTHREADS];
    __shared__ int   s_idx[NTHREADS];

    const int b   = blockIdx.x;
    const int tid = threadIdx.x;
    const int p   = pma[b];
    const int lo  = p - 4;
    const int hi  = p + 3;
    float* e_row  = alignments + (size_t)b * T_;

    // pass 1: masked max + first-occurrence argmax
    float m = NEGV;
    int   mi = T_;
    for (int t = tid; t < T_; t += NTHREADS) {
        float e = (t >= lo && t < hi) ? e_row[t] : NEGV;
        if (e > m || (e == m && t < mi)) { m = e; mi = t; }
    }
    s_red[tid] = m; s_idx[tid] = mi;
    __syncthreads();
    for (int s = NTHREADS / 2; s > 0; s >>= 1) {
        if (tid < s) {
            float om = s_red[tid + s]; int oi = s_idx[tid + s];
            if (om > s_red[tid] || (om == s_red[tid] && oi < s_idx[tid])) {
                s_red[tid] = om; s_idx[tid] = oi;
            }
        }
        __syncthreads();
    }
    const float rmax = s_red[0];
    const int   amax = s_idx[0];
    __syncthreads();

    // pass 2: sum of exp (masked entries underflow to exactly 0)
    float sum = 0.0f;
    for (int t = tid; t < T_; t += NTHREADS) {
        float e = (t >= lo && t < hi) ? e_row[t] : NEGV;
        sum += expf(e - rmax);
    }
    s_red[tid] = sum;
    __syncthreads();
    for (int s = NTHREADS / 2; s > 0; s >>= 1) {
        if (tid < s) s_red[tid] += s_red[tid + s];
        __syncthreads();
    }
    const float inv = 1.0f / s_red[0];
    __syncthreads();

    // pass 3: in-place alignments + next_state (thread-local read-then-write)
    const float* st_row = state + (size_t)b * T_;
    for (int t = tid; t < T_; t += NTHREADS) {
        float e = (t >= lo && t < hi) ? e_row[t] : NEGV;
        float a = expf(e - rmax) * inv;
        e_row[t] = a;
        next_state[(size_t)b * T_ + t] = a + st_row[t];
    }
    if (tid == 0) max_att[b] = amax;
}

// ---------------------------------------------------------------------------
extern "C" void kernel_launch(void* const* d_in, const int* in_sizes, int n_in,
                              void* d_out, int out_size, void* d_ws, size_t ws_size,
                              hipStream_t stream)
{
    (void)in_sizes; (void)n_in; (void)out_size; (void)d_ws; (void)ws_size;

    const float* query  = (const float*)d_in[0];
    const float* state  = (const float*)d_in[1];
    const float* memory = (const float*)d_in[2];
    const float* conv_w = (const float*)d_in[3];
    const float* conv_b = (const float*)d_in[4];
    const float* loc_w  = (const float*)d_in[5];
    const float* v_a    = (const float*)d_in[6];
    const float* b_a    = (const float*)d_in[7];
    const int*   pma    = (const int*)d_in[8];

    float* alignments = (float*)d_out;                         // B*T
    float* next_state = alignments + (size_t)B_ * T_;          // B*T
    int*   max_att    = (int*)(next_state + (size_t)B_ * T_);  // B (int32 bits)

    dim3 g1(B_, T_ / TBLOCK);
    lsa_energy_kernel<<<g1, NTHREADS, 0, stream>>>(
        query, state, memory, conv_w, conv_b, loc_w, v_a, b_a, alignments);

    lsa_softmax_kernel<<<B_, NTHREADS, 0, stream>>>(
        state, pma, alignments, next_state, max_att);
}
